// SS2D_36026185679089
// MI455X (gfx1250) — matrix operand order repaired and checked by reading, not compiled, not run on hardware
//
#include <hip/hip_runtime.h>
#include <hip/hip_bf16.h>

// ---------------------------------------------------------------------------
// SS2D forward for gfx1250 (MI455X), wave32 + WMMA f32_16x16x32_f16 +
// async global->LDS staging (ASYNCcnt path).
// Pipeline:
//   1) k_in_proj : xz = x @ W_in^T  (WMMA)  -> xm, z=silu(z)
//   2) k_conv    : depthwise 3x3 + bias + silu -> xc  (B,L,D) layout
//   3) k_xproj   : x_dbl[b,k,c,l] = sum_d xs * x_proj_w (WMMA; weight tile
//                  staged in LDS via global_load_async_to_lds_b128; xs
//                  gathered from xc via direction index map, never
//                  materialized)
//   4) k_scan    : sequential selective scan, 4 directions -> bufA/bufB
//   5) k_ln      : combine directions + LayerNorm + *z -> ypre
//   6) k_out_proj: out = ypre @ W_out^T (WMMA)
// ---------------------------------------------------------------------------

typedef _Float16 v16h __attribute__((ext_vector_type(16)));
typedef float    v8f  __attribute__((ext_vector_type(8)));

#define HIMG   96
#define WIMG   96
#define LSEQ   9216           // 96*96
#define NBATCH 4
#define DMODEL 96
#define DINNER 192
#define DSTATE 16
#define DTRANK 6
#define KDIR   4
#define CDBL   38             // DT_RANK + 2*D_STATE
#define NROWS  (NBATCH*LSEQ)  // 36864

// fast silu: one v_exp_f32 + one v_rcp_f32 (avoid IEEE div expansion)
__device__ __forceinline__ float silu_f(float t) {
  return t * __builtin_amdgcn_rcpf(1.f + __expf(-t));
}

// direction index map: scan step i of direction k reads/writes memory
// position m (in row-major h*W+w order).  k0: i, k1: transpose, k2: flip,
// k3: flip+transpose.  u-source and combined-y target share the same index.
__device__ __forceinline__ int dir_map(int k, int i) {
  int t = (k & 2) ? (LSEQ - 1 - i) : i;
  if (k & 1) t = (t % 96) * 96 + t / 96;
  return t;
}

// ---- async global->LDS copy (CDNA5 ASYNCcnt path) --------------------------
// ISA 10.2: generic address of an LDS location truncates to the LDS byte
// offset, which is exactly what the VDST operand of the async load wants.
__device__ __forceinline__ void async_copy_b128_to_lds(void* lds_dst,
                                                       const void* gsrc) {
  unsigned lds_off = (unsigned)(unsigned long long)lds_dst;
  asm volatile("global_load_async_to_lds_b128 %0, %1, off"
               :: "v"(lds_off), "v"(gsrc) : "memory");
}
__device__ __forceinline__ void wait_async_lds() {
  asm volatile("s_wait_asynccnt 0x0" ::: "memory");
}

// ---- WMMA fragment loaders (contiguous k order, f16 16x16x32) --------------
// A (16xK tile, row-major fp32 source): lane r(0..15)=row M in both halves;
// half 0 holds K = kb+0..15, half 1 holds kb+16..31.
__device__ __forceinline__ v16h load_a_frag(const float* rowA, int kb, int half_id) {
  const float* p = rowA + kb + 16 * half_id;
  float4 f0 = *(const float4*)(p);
  float4 f1 = *(const float4*)(p + 4);
  float4 f2 = *(const float4*)(p + 8);
  float4 f3 = *(const float4*)(p + 12);
  v16h a;
  a[0]=(_Float16)f0.x;  a[1]=(_Float16)f0.y;  a[2]=(_Float16)f0.z;  a[3]=(_Float16)f0.w;
  a[4]=(_Float16)f1.x;  a[5]=(_Float16)f1.y;  a[6]=(_Float16)f1.z;  a[7]=(_Float16)f1.w;
  a[8]=(_Float16)f2.x;  a[9]=(_Float16)f2.y;  a[10]=(_Float16)f2.z; a[11]=(_Float16)f2.w;
  a[12]=(_Float16)f3.x; a[13]=(_Float16)f3.y; a[14]=(_Float16)f3.z; a[15]=(_Float16)f3.w;
  return a;
}
// B (Kx16 tile): lane r = column N; half 0 holds K = kb+0..15, half 1
// K = kb+16..31.  Source row = weight row for column N (contiguous in K).
__device__ __forceinline__ v16h load_b_frag(const float* rowB, int kb, int half_id) {
  const float* p = rowB + kb + 16 * half_id;
  float4 f0 = *(const float4*)(p);
  float4 f1 = *(const float4*)(p + 4);
  float4 f2 = *(const float4*)(p + 8);
  float4 f3 = *(const float4*)(p + 12);
  v16h b;
  b[0]=(_Float16)f0.x;  b[1]=(_Float16)f0.y;  b[2]=(_Float16)f0.z;  b[3]=(_Float16)f0.w;
  b[4]=(_Float16)f1.x;  b[5]=(_Float16)f1.y;  b[6]=(_Float16)f1.z;  b[7]=(_Float16)f1.w;
  b[8]=(_Float16)f2.x;  b[9]=(_Float16)f2.y;  b[10]=(_Float16)f2.z; b[11]=(_Float16)f2.w;
  b[12]=(_Float16)f3.x; b[13]=(_Float16)f3.y; b[14]=(_Float16)f3.z; b[15]=(_Float16)f3.w;
  return b;
}

// ---- 1) input projection: (36864 x 96) @ (384 x 96)^T ----------------------
__global__ __launch_bounds__(256) void k_in_proj(const float* __restrict__ X,
    const float* __restrict__ Win, float* __restrict__ xm, float* __restrict__ zbuf)
{
  int wave = (blockIdx.x * 256 + threadIdx.x) >> 5;   // 55296 waves
  int mt = wave / 24, nt = wave % 24;
  int lane = threadIdx.x & 31;
  int half_id = lane >> 4, r = lane & 15;
  int m0 = mt * 16, n0 = nt * 16;
  const float* rowA = X   + (m0 + r) * DMODEL;
  const float* rowB = Win + (n0 + r) * DMODEL;
  v8f acc = {};
  #pragma unroll
  for (int kb = 0; kb < DMODEL; kb += 32) {
    v16h af = load_a_frag(rowA, kb, half_id);
    v16h bf = load_b_frag(rowB, kb, half_id);
    acc = __builtin_amdgcn_wmma_f32_16x16x32_f16(false, af, false, bf,
                                                 (short)0, acc, false, false);
  }
  int n = n0 + r;
  int mrow = m0 + 8 * half_id;    // D frag: half0 rows 0..7, half1 rows 8..15
  if (n < DINNER) {
    #pragma unroll
    for (int v = 0; v < 8; ++v) xm[(mrow + v) * DINNER + n] = acc[v];
  } else {
    int nz = n - DINNER;
    #pragma unroll
    for (int v = 0; v < 8; ++v)
      zbuf[(mrow + v) * DINNER + nz] = silu_f(acc[v]);
  }
}

// ---- 2) depthwise 3x3 conv + bias + silu -----------------------------------
__global__ __launch_bounds__(256) void k_conv(const float* __restrict__ xm,
    const float* __restrict__ cw, const float* __restrict__ cb,
    float* __restrict__ xc)
{
  int idx = blockIdx.x * 256 + threadIdx.x;
  if (idx >= NROWS * DINNER) return;
  int d = idx % DINNER;
  int l = (idx / DINNER) % LSEQ;
  int b = idx / (DINNER * LSEQ);
  int h = l / WIMG, w = l % WIMG;
  float s = cb[d];
  #pragma unroll
  for (int dh = -1; dh <= 1; ++dh) {
    #pragma unroll
    for (int dw = -1; dw <= 1; ++dw) {
      int hh = h + dh, ww = w + dw;
      if (hh >= 0 && hh < HIMG && ww >= 0 && ww < WIMG)
        s += xm[(b * LSEQ + hh * WIMG + ww) * DINNER + d] *
             cw[d * 9 + (dh + 1) * 3 + (dw + 1)];
    }
  }
  xc[idx] = silu_f(s);
}

// ---- 3) x_proj: x_dbl[b,k,c,l] = sum_d xs[b,k,d,l] * xpw[k,c,d] (WMMA) -----
// 8-wave blocks; all waves of a block share one (b,k), so the 38x192 f32
// weight tile (28.5 KB) is staged once per block into LDS with async loads.
__global__ __launch_bounds__(256) void k_xproj(const float* __restrict__ xc,
    const float* __restrict__ xpw, float* __restrict__ xdbl)
{
  __shared__ float sB[CDBL * DINNER];          // 38*192 floats = 29184 B

  // stage weight tile for this block's k (uniform across the block)
  {
    int k_blk = ((blockIdx.x * 8) / (576 * 3)) & 3;
    const float* src = xpw + k_blk * CDBL * DINNER;
    for (int it = threadIdx.x; it < (CDBL * DINNER) / 4; it += 256)
      async_copy_b128_to_lds(&sB[it * 4], src + it * 4);
    wait_async_lds();                           // s_wait_asynccnt 0
    __syncthreads();
  }

  int wave = (blockIdx.x * 256 + threadIdx.x) >> 5;   // 27648 waves
  int nt = wave % 3;
  int tmp = wave / 3;
  int lt = tmp % 576;
  int bk = tmp / 576;
  int b = bk >> 2, k = bk & 3;
  int lane = threadIdx.x & 31;
  int half_id = lane >> 4, r = lane & 15;
  int l0 = lt * 16;
  int c  = nt * 16 + r;                    // output channel (N), 38 valid
  bool cvalid = c < CDBL;
  const float* rowB = sB + (cvalid ? c : 0) * DINNER;   // LDS reads
  const float* rowA = xc + (b * LSEQ + dir_map(k, l0 + r)) * DINNER;  // xs row
  v8f acc = {};
  #pragma unroll
  for (int kb = 0; kb < DINNER; kb += 32) {
    v16h af = load_a_frag(rowA, kb, half_id);
    v16h bf = load_b_frag(rowB, kb, half_id);
    if (!cvalid) {
      #pragma unroll
      for (int j = 0; j < 16; ++j) bf[j] = (_Float16)0.f;
    }
    acc = __builtin_amdgcn_wmma_f32_16x16x32_f16(false, af, false, bf,
                                                 (short)0, acc, false, false);
  }
  if (cvalid) {
    int base = (bk * CDBL + c) * LSEQ + l0 + 8 * half_id;  // 8 contiguous l
    #pragma unroll
    for (int v = 0; v < 8; ++v) xdbl[base + v] = acc[v];
  }
}

// ---- 4) selective scan (sequential in L), 2 direction-pairs ----------------
// block = 256 threads = 16 d-channels x 16 states; grid = B * 12 * 2 pairs.
// Each thread owns one (b,k-pair,d,n); y-reduction over n via width-16 xor
// shuffles; the second direction of a pair accumulates into the same buffer
// (same owning thread -> deterministic, no atomics).
__global__ __launch_bounds__(256) void k_scan(const float* __restrict__ xc,
    const float* __restrict__ xdbl, const float* __restrict__ dtw,
    const float* __restrict__ dtb, const float* __restrict__ Alog,
    const float* __restrict__ Dsp, float* __restrict__ bufA,
    float* __restrict__ bufB)
{
  int blk = blockIdx.x;
  int pair = blk & 1;                 // 0 -> k in {0,2}; 1 -> k in {1,3}
  int dchunk = (blk >> 1) % 12;
  int b = blk / 24;
  int tid = threadIdx.x;
  int n = tid & 15;
  int d = dchunk * 16 + (tid >> 4);
  float* buf = pair ? bufB : bufA;
  #pragma unroll 1
  for (int kk = 0; kk < 2; ++kk) {
    int k = pair + 2 * kk;
    int kd = k * DINNER + d;
    float a_n  = -__expf(Alog[kd * DSTATE + n]);
    float dsv  = Dsp[kd];
    float bias = dtb[kd];
    float w0 = dtw[kd*DTRANK+0], w1 = dtw[kd*DTRANK+1], w2 = dtw[kd*DTRANK+2];
    float w3 = dtw[kd*DTRANK+3], w4 = dtw[kd*DTRANK+4], w5 = dtw[kd*DTRANK+5];
    const float* xd  = xdbl + (b * KDIR + k) * CDBL * LSEQ;
    const float* xdB = xd + (DTRANK + n) * LSEQ;
    const float* xdC = xd + (DTRANK + DSTATE + n) * LSEQ;
    float h = 0.f;
    for (int i = 0; i < LSEQ; ++i) {
      int m = dir_map(k, i);
      float u = xc[(b * LSEQ + m) * DINNER + d];
      float dtraw = bias + w0 * xd[i]            + w1 * xd[LSEQ + i]
                         + w2 * xd[2 * LSEQ + i] + w3 * xd[3 * LSEQ + i]
                         + w4 * xd[4 * LSEQ + i] + w5 * xd[5 * LSEQ + i];
      float delta = (dtraw > 20.f) ? dtraw : log1pf(__expf(dtraw));  // softplus
      float Bn = xdB[i];
      float Cn = xdC[i];
      h = h * __expf(delta * a_n) + (delta * u) * Bn;
      float p = h * Cn;
      p += __shfl_xor(p, 1, 16);
      p += __shfl_xor(p, 2, 16);
      p += __shfl_xor(p, 4, 16);
      p += __shfl_xor(p, 8, 16);
      if (n == 0) {
        float y = p + dsv * u;                 // + xs * Ds term
        int o = (b * LSEQ + m) * DINNER + d;
        buf[o] = kk ? (buf[o] + y) : y;
      }
      if (i + 64 < LSEQ) {                     // global_prefetch_b8
        __builtin_prefetch(&xd[i + 64], 0, 1);
        __builtin_prefetch(&xc[(b * LSEQ + dir_map(k, i + 64)) * DINNER + d], 0, 1);
      }
    }
  }
}

// ---- 5) combine directions + LayerNorm + gate by z -------------------------
__global__ __launch_bounds__(256) void k_ln(const float* __restrict__ bufA,
    const float* __restrict__ bufB, const float* __restrict__ zbuf,
    const float* __restrict__ lnw, const float* __restrict__ lnb,
    float* __restrict__ ypre)
{
  int pos  = (blockIdx.x * 256 + threadIdx.x) >> 5;  // one wave per (b,l)
  int lane = threadIdx.x & 31;
  const float* pa = bufA + pos * DINNER;
  const float* pb = bufB + pos * DINNER;
  float vals[6];
  float s = 0.f;
  #pragma unroll
  for (int c2 = 0; c2 < 6; ++c2) {
    int d = c2 * 32 + lane;
    float v = pa[d] + pb[d];
    vals[c2] = v;
    s += v;
  }
  #pragma unroll
  for (int off = 1; off < 32; off <<= 1) s += __shfl_xor(s, off, 32);
  float mu = s * (1.f / 192.f);
  float s2 = 0.f;
  #pragma unroll
  for (int c2 = 0; c2 < 6; ++c2) { float t = vals[c2] - mu; s2 += t * t; }
  #pragma unroll
  for (int off = 1; off < 32; off <<= 1) s2 += __shfl_xor(s2, off, 32);
  float inv = rsqrtf(s2 * (1.f / 192.f) + 1e-5f);
  #pragma unroll
  for (int c2 = 0; c2 < 6; ++c2) {
    int d = c2 * 32 + lane;
    float yv = (vals[c2] - mu) * inv * lnw[d] + lnb[d];
    ypre[pos * DINNER + d] = yv * zbuf[pos * DINNER + d];
  }
}

// ---- 6) output projection: (36864 x 192) @ (96 x 192)^T --------------------
__global__ __launch_bounds__(256) void k_out_proj(const float* __restrict__ Y,
    const float* __restrict__ Wout, float* __restrict__ out)
{
  int wave = (blockIdx.x * 256 + threadIdx.x) >> 5;   // 13824 waves
  int mt = wave / 6, nt = wave % 6;
  int lane = threadIdx.x & 31;
  int half_id = lane >> 4, r = lane & 15;
  int m0 = mt * 16, n0 = nt * 16;
  const float* rowA = Y    + (m0 + r) * DINNER;
  const float* rowB = Wout + (n0 + r) * DINNER;
  v8f acc = {};
  #pragma unroll
  for (int kb = 0; kb < DINNER; kb += 32) {
    v16h af = load_a_frag(rowA, kb, half_id);
    v16h bf = load_b_frag(rowB, kb, half_id);
    acc = __builtin_amdgcn_wmma_f32_16x16x32_f16(false, af, false, bf,
                                                 (short)0, acc, false, false);
  }
  int n = n0 + r;
  int mrow = m0 + 8 * half_id;
  #pragma unroll
  for (int v = 0; v < 8; ++v) out[(mrow + v) * DMODEL + n] = acc[v];
}

// ---------------------------------------------------------------------------
extern "C" void kernel_launch(void* const* d_in, const int* in_sizes, int n_in,
                              void* d_out, int out_size, void* d_ws, size_t ws_size,
                              hipStream_t stream)
{
  (void)in_sizes; (void)n_in; (void)out_size; (void)ws_size;
  const float* x    = (const float*)d_in[0];
  const float* Win  = (const float*)d_in[1];
  const float* cw   = (const float*)d_in[2];
  const float* cb   = (const float*)d_in[3];
  const float* xpw  = (const float*)d_in[4];
  const float* dtw  = (const float*)d_in[5];
  const float* dtb  = (const float*)d_in[6];
  const float* Alog = (const float*)d_in[7];
  const float* Dsp  = (const float*)d_in[8];
  const float* lnw  = (const float*)d_in[9];
  const float* lnb  = (const float*)d_in[10];
  const float* Wout = (const float*)d_in[11];

  // workspace carve-out (floats): 6*7,077,888 + 5,603,328 = ~192.3 MB
  const size_t NE = (size_t)NROWS * DINNER;            // 7,077,888
  float* ws   = (float*)d_ws;
  float* xm   = ws;
  float* zbuf = xm + NE;
  float* xc   = zbuf + NE;
  float* xdbl = xc + NE;                               // 16*38*9216
  float* bufA = xdbl + (size_t)NBATCH * KDIR * CDBL * LSEQ;
  float* bufB = bufA + NE;
  float* ypre = bufB + NE;
  float* outp = (float*)d_out;

  k_in_proj <<<6912, 256, 0, stream>>>(x, Win, xm, zbuf);
  k_conv    <<<27648, 256, 0, stream>>>(xm, cw, cb, xc);
  k_xproj   <<<3456, 256, 0, stream>>>(xc, xpw, xdbl);
  k_scan    <<<96,   256, 0, stream>>>(xc, xdbl, dtw, dtb, Alog, Dsp, bufA, bufB);
  k_ln      <<<4608, 256, 0, stream>>>(bufA, bufB, zbuf, lnw, lnb, ypre);
  k_out_proj<<<1728, 256, 0, stream>>>(ypre, Wout, outp);
}
